// G_QNet_86749749444734
// MI455X (gfx1250) — matrix operand order repaired and checked by reading, not compiled, not run on hardware
//
#include <hip/hip_runtime.h>
#include <hip/hip_bf16.h>

typedef __attribute__((ext_vector_type(16))) _Float16 v16h;
typedef __attribute__((ext_vector_type(8)))  float    v8f;

#define B_      4096
#define KNN_    8
#define N_      256     // nodes per graph
#define E_      4096    // edges per graph
#define PER_AG  322     // FROM*NDD + TO*NDD + 2
#define XROW    (KNN_*PER_AG)   // 2576
#define MID_    3
#define AFTER_  2
#define COMMON_ 6
#define R_      6
#define NDE_    4
#define D1_     1024
#define KPAD    320     // 304 padded to 10 * 32
#define KSTEPS  10
#define NTILES  64      // 1024 / 16
#define A_      5
#define EPS_    1e-5f

__device__ __forceinline__ float fsgnsqrt(float t) {
    return copysignf(sqrtf(fabsf(t)), t);
}

// Async bulk copy: 16 bytes global -> LDS, tracked by ASYNCcnt (CDNA5 path).
// lds_dst: flat pointer into __shared__ (low 32 bits == LDS byte address).
// gbase:   uniform global base (SGPR pair); byte_off: per-lane byte offset.
__device__ __forceinline__ void async_copy_b128(void* lds_dst,
                                                const void* gbase,
                                                unsigned byte_off) {
    unsigned lds = (unsigned)(unsigned long long)lds_dst;
    asm volatile("global_load_async_to_lds_b128 %0, %1, %2"
                 :: "v"(lds), "v"(byte_off), "s"(gbase)
                 : "memory");
}
__device__ __forceinline__ void wait_async_done() {
    asm volatile("s_wait_asynccnt 0x0" ::: "memory");
}

__device__ __forceinline__ float block_sum(float v, float* red, int tid) {
    red[tid] = v; __syncthreads();
    #pragma unroll
    for (int off = 128; off > 0; off >>= 1) {
        if (tid < off) red[tid] += red[tid + off];
        __syncthreads();
    }
    float r = red[0];
    __syncthreads();
    return r;
}

// ---------------------------------------------------------------------------
// Kernel A: per-graph GCN pipeline entirely in LDS, emits fp16 feat rows.
// One block per batch element; thread n owns node n. Edge slab (32 KB) is
// async-copied to LDS once and re-read three times from LDS.
// ---------------------------------------------------------------------------
__global__ __launch_bounds__(256)
void graph_feat_kernel(const float* __restrict__ x,
                       const int*   __restrict__ ledge,
                       const int*   __restrict__ rows_common,
                       const float* __restrict__ each_info,
                       const float* __restrict__ gw1, const float* __restrict__ gb1,
                       const float* __restrict__ gw3, const float* __restrict__ gb3,
                       const float* __restrict__ lw1, const float* __restrict__ lb1,
                       const float* __restrict__ lw3, const float* __restrict__ lb3,
                       _Float16* __restrict__ feat16)
{
    __shared__ int   edgeS[2 * E_];          // 32 KB: [0..E) = src, [E..2E) = dst
    __shared__ float nodesS[N_][COMMON_];
    __shared__ float hS[N_][MID_];
    __shared__ float oS[N_][MID_];
    __shared__ float dS[N_];
    __shared__ float red[256];
    __shared__ float h2S[N_][AFTER_];
    __shared__ float o2S[N_][AFTER_];

    const int b   = blockIdx.x;
    const int tid = threadIdx.x;
    const int n   = tid;
    const int k   = n >> 5;     // agent index 0..7
    const int i   = n & 31;     // node-in-agent 0..31

    const float* xb = x + (size_t)b * XROW;

    // kick off async edge-slab copy (2*E ints = 2048 x 16B chunks)
    {
        const void* ebase = (const void*)(ledge + (size_t)b * 2 * E_);
        for (int c = tid; c < (2 * E_ * 4) / 16; c += 256)
            async_copy_b128((char*)edgeS + c * 16, ebase, (unsigned)(c * 16));
    }

    // overlap: assemble node features while the DMA runs
    // nodes = concat(x_start[...,4:10], x_dest[...,4:10]) -> (256, 6)
    const int base = k * PER_AG + ((i < 16) ? (i * 10 + 4)
                                            : (160 + (i - 16) * 10 + 4));
    #pragma unroll
    for (int c = 0; c < COMMON_; ++c) nodesS[n][c] = xb[base + c];
    dS[n] = 1.0f;

    wait_async_done();
    __syncthreads();

    const int* srcp = edgeS;
    const int* dstp = edgeS + E_;

    // degree accumulation (LDS float atomics -> ds_add_f32)
    for (int e = tid; e < E_; e += 256)
        atomicAdd(&dS[dstp[e] & (N_ - 1)], 1.0f);
    __syncthreads();

    dS[n] = rsqrtf(dS[n]);          // d = rsqrt(deg), reused by both layers

    // ---- GCN layer 1: 6 -> 3 ----
    #pragma unroll
    for (int m = 0; m < MID_; ++m) {
        float s = 0.f;
        #pragma unroll
        for (int c = 0; c < COMMON_; ++c) s += nodesS[n][c] * gw1[c * MID_ + m];
        hS[n][m] = s;
    }
    __syncthreads();

    const float dn = dS[n];
    #pragma unroll
    for (int m = 0; m < MID_; ++m) oS[n][m] = hS[n][m] * dn * dn + gb1[m];
    __syncthreads();

    for (int e = tid; e < E_; e += 256) {
        int s = srcp[e] & (N_ - 1), d = dstp[e] & (N_ - 1);
        float coef = dS[s] * dS[d];
        #pragma unroll
        for (int m = 0; m < MID_; ++m)
            atomicAdd(&oS[d][m], hS[s][m] * coef);
    }
    __syncthreads();

    // graph-wide LayerNorm over 256*3 values, then ReLU -> hS
    {
        float lsum = 0.f, lsq = 0.f;
        #pragma unroll
        for (int m = 0; m < MID_; ++m) { float v = oS[n][m]; lsum += v; lsq += v * v; }
        float S  = block_sum(lsum, red, tid);
        float Q  = block_sum(lsq,  red, tid);
        float m0 = S * (1.f / (N_ * MID_));
        float vr = Q * (1.f / (N_ * MID_)) - m0 * m0;
        float rs = rsqrtf(vr + EPS_);
        #pragma unroll
        for (int m = 0; m < MID_; ++m)
            hS[n][m] = fmaxf((oS[n][m] - m0) * rs * lw1[m] + lb1[m], 0.f);
    }
    __syncthreads();

    // ---- GCN layer 2: 3 -> 2 ----
    #pragma unroll
    for (int q = 0; q < AFTER_; ++q) {
        float s = 0.f;
        #pragma unroll
        for (int m = 0; m < MID_; ++m) s += hS[n][m] * gw3[m * AFTER_ + q];
        h2S[n][q] = s;
    }
    __syncthreads();

    #pragma unroll
    for (int q = 0; q < AFTER_; ++q) o2S[n][q] = h2S[n][q] * dn * dn + gb3[q];
    __syncthreads();

    for (int e = tid; e < E_; e += 256) {
        int s = srcp[e] & (N_ - 1), d = dstp[e] & (N_ - 1);
        float coef = dS[s] * dS[d];
        #pragma unroll
        for (int q = 0; q < AFTER_; ++q)
            atomicAdd(&o2S[d][q], h2S[s][q] * coef);
    }
    __syncthreads();

    {
        float lsum = 0.f, lsq = 0.f;
        #pragma unroll
        for (int q = 0; q < AFTER_; ++q) { float v = o2S[n][q]; lsum += v; lsq += v * v; }
        float S  = block_sum(lsum, red, tid);
        float Q  = block_sum(lsq,  red, tid);
        float m0 = S * (1.f / (N_ * AFTER_));
        float vr = Q * (1.f / (N_ * AFTER_)) - m0 * m0;
        float rs = rsqrtf(vr + EPS_);
        #pragma unroll
        for (int q = 0; q < AFTER_; ++q)
            o2S[n][q] = fmaxf((o2S[n][q] - m0) * rs * lw3[q] + lb3[q], 0.f);
    }
    __syncthreads();

    // ---- gather + feature assembly: feat row = 8 agents x 38, pad to 320 ----
    _Float16*    fout = feat16 + (size_t)b * KPAD;
    const int*   rc   = rows_common + (size_t)b * KNN_ * R_;
    const float* ei   = each_info + (size_t)b * KNN_ * R_ * NDE_;

    for (int j = tid; j < KPAD; j += 256) {
        float val = 0.f;
        if (j < KNN_ * 38) {
            int kk = j / 38, jj = j % 38;
            if (jj < 12) {                       // gathered GCN output
                int r = jj >> 1, a = jj & 1;
                int node = rc[kk * R_ + r] & (N_ - 1);
                val = o2S[node][a];
            } else if (jj < 14) {                // x_else
                val = xb[kk * PER_AG + 320 + (jj - 12)];
            } else {                             // each_info flat
                val = ei[kk * R_ * NDE_ + (jj - 14)];
            }
        }
        fout[j] = (_Float16)val;
    }
}

// ---------------------------------------------------------------------------
// Kernel W: pack l1_w (1024 x 304, fp32) into WMMA B-fragment order (fp16).
// Fragment layout (32x16 K x N tile): N = lane%16, half index = K (table order).
// ---------------------------------------------------------------------------
__global__ __launch_bounds__(32)
void pack_w_kernel(const float* __restrict__ l1w, _Float16* __restrict__ bp)
{
    const int lane  = threadIdx.x;       // N = lane%16, K half = lane/16
    const int kstep = blockIdx.x >> 6;   // 0..9
    const int ntile = blockIdx.x & 63;   // 0..63
    const int kb = kstep * 32 + ((lane < 16) ? 0 : 8);
    _Float16* out = bp + ((size_t)(kstep * NTILES + ntile) * 32 + lane) * 16;
    #pragma unroll
    for (int h = 0; h < 16; ++h) {
        int nn = ntile * 16 + (lane & 15);
        int K = kb + ((h < 8) ? h : h + 8);
        float v = (K < 304) ? l1w[(size_t)nn * 304 + K] : 0.f;
        out[h] = (_Float16)v;
    }
}

// ---------------------------------------------------------------------------
// Kernel G: WMMA GEMM (16-row strip per block) fused with bias + row-LN +
// ReLU + noisy dueling heads. 8 waves x 8 N-tiles x 10 K-steps of
// v_wmma_f32_16x16x32_f16. A strip async-copied to LDS.
// ---------------------------------------------------------------------------
__global__ __launch_bounds__(256)
void gemm_head_kernel(const _Float16* __restrict__ feat16,
                      const _Float16* __restrict__ bpack,
                      const float* __restrict__ l1b,
                      const float* __restrict__ ln1w, const float* __restrict__ ln1b,
                      const float* __restrict__ adv_uw, const float* __restrict__ adv_sw,
                      const float* __restrict__ adv_ub, const float* __restrict__ adv_sb,
                      const float* __restrict__ v_uw,  const float* __restrict__ v_sw,
                      const float* __restrict__ v_ub,  const float* __restrict__ v_sb,
                      const float* __restrict__ adv_rin, const float* __restrict__ adv_rout,
                      const float* __restrict__ v_rin,  const float* __restrict__ v_rout,
                      float* __restrict__ out)
{
    __shared__ _Float16 As[16][KPAD];    // 10 KB, strip is contiguous in global
    __shared__ _Float16 hidS[16][D1_];   // 32 KB

    const int tid  = threadIdx.x;
    const int lane = tid & 31;
    const int w    = tid >> 5;
    const int row0 = blockIdx.x * 16;

    // async-stage A strip (16 x 320 fp16 = 10240 B = 640 x 16B chunks)
    {
        const void* abase = (const void*)(feat16 + (size_t)row0 * KPAD);
        for (int c = tid; c < (16 * KPAD * 2) / 16; c += 256)
            async_copy_b128((char*)As + c * 16, abase, (unsigned)(c * 16));
    }
    wait_async_done();
    __syncthreads();

    v8f acc[8] = {};
    const int m    = lane & 15;
    const int koff = (lane < 16) ? 0 : 8;
    const v16h* bp = (const v16h*)bpack;

    for (int ks = 0; ks < KSTEPS; ++ks) {
        const int k0 = ks * 32;
        // A fragment per documented 16-bit 16x32 layout:
        // lanes 0-15:  M=lane, K = {0..7, 16..23};  lanes 16-31: M=lane-16, K = {8..15, 24..31}
        v16h a;
        #pragma unroll
        for (int ii = 0; ii < 8; ++ii) a[ii]     = As[m][k0 + koff + ii];
        #pragma unroll
        for (int ii = 0; ii < 8; ++ii) a[8 + ii] = As[m][k0 + 16 + koff + ii];

        #pragma unroll
        for (int t = 0; t < 8; ++t) {
            v16h bf = bp[(size_t)(ks * NTILES + (w * 8 + t)) * 32 + lane];
            acc[t] = __builtin_amdgcn_wmma_f32_16x16x32_f16(
                false, a, false, bf, (short)0, acc[t], false, false);
        }
        if (ks + 1 < KSTEPS)
            __builtin_prefetch(&bpack[(size_t)((ks + 1) * NTILES + w * 8) * 32 * 16], 0, 1);
    }

    // C layout: VGPR r -> M=r (lanes 0-15) / M=r+8 (lanes 16-31), N = lane&15
    #pragma unroll
    for (int t = 0; t < 8; ++t) {
        const int col   = (w * 8 + t) * 16 + (lane & 15);
        const int rbase = (lane < 16) ? 0 : 8;
        #pragma unroll
        for (int r = 0; r < 8; ++r)
            hidS[rbase + r][col] = (_Float16)(acc[t][r] + l1b[col]);
    }
    __syncthreads();

    // epilogue: wave w owns rows 2w and 2w+1
    float fo[A_];
    #pragma unroll
    for (int a = 0; a < A_; ++a) fo[a] = fsgnsqrt(adv_rout[a]);
    const float fv = fsgnsqrt(v_rout[0]);

    for (int rr = 0; rr < 2; ++rr) {
        const int row = w * 2 + rr;

        // row mean / variance (1024 values, 32 per lane)
        float sum = 0.f, sq = 0.f;
        for (int ii = 0; ii < 32; ++ii) {
            float v = (float)hidS[row][lane + ii * 32];
            sum += v; sq += v * v;
        }
        #pragma unroll
        for (int off = 16; off > 0; off >>= 1) {
            sum += __shfl_xor(sum, off);
            sq  += __shfl_xor(sq,  off);
        }
        const float mm = sum * (1.f / D1_);
        const float vr = sq  * (1.f / D1_) - mm * mm;
        const float rs = rsqrtf(vr + EPS_);

        // LN + ReLU fused into the 6 noisy-head dot products
        float dacc[A_ + 1] = {0.f, 0.f, 0.f, 0.f, 0.f, 0.f};
        for (int ii = 0; ii < 32; ++ii) {
            const int c = lane + ii * 32;
            float hv = ((float)hidS[row][c] - mm) * rs * ln1w[c] + ln1b[c];
            hv = fmaxf(hv, 0.f);
            const float fin  = fsgnsqrt(adv_rin[c]);
            const float fvin = fsgnsqrt(v_rin[c]);
            #pragma unroll
            for (int a = 0; a < A_; ++a)
                dacc[a] += hv * (adv_uw[a * D1_ + c] + adv_sw[a * D1_ + c] * fo[a] * fin);
            dacc[A_] += hv * (v_uw[c] + v_sw[c] * fv * fvin);
        }
        #pragma unroll
        for (int j = 0; j < A_ + 1; ++j) {
            #pragma unroll
            for (int off = 16; off > 0; off >>= 1)
                dacc[j] += __shfl_xor(dacc[j], off);
        }

        if (lane == 0) {
            float adv[A_], amean = 0.f;
            #pragma unroll
            for (int a = 0; a < A_; ++a) {
                adv[a] = dacc[a] + adv_ub[a] + adv_sb[a] * fo[a];
                amean += adv[a];
            }
            amean *= (1.f / A_);
            const float vv = dacc[A_] + v_ub[0] + v_sb[0] * fv;
            const int gr = row0 + row;
            #pragma unroll
            for (int a = 0; a < A_; ++a)
                out[(size_t)gr * A_ + a] = vv + adv[a] - amean;
        }
    }
}

// ---------------------------------------------------------------------------
extern "C" void kernel_launch(void* const* d_in, const int* in_sizes, int n_in,
                              void* d_out, int out_size, void* d_ws, size_t ws_size,
                              hipStream_t stream) {
    const float* x        = (const float*)d_in[0];
    const int*   ledge    = (const int*)  d_in[1];
    const int*   rows     = (const int*)  d_in[2];
    const float* einfo    = (const float*)d_in[3];
    const float* gcn1_w   = (const float*)d_in[4];
    const float* gcn1_b   = (const float*)d_in[5];
    const float* gcn3_w   = (const float*)d_in[6];
    const float* gcn3_b   = (const float*)d_in[7];
    const float* lng1_w   = (const float*)d_in[8];
    const float* lng1_b   = (const float*)d_in[9];
    const float* lng3_w   = (const float*)d_in[10];
    const float* lng3_b   = (const float*)d_in[11];
    const float* l1_w     = (const float*)d_in[12];
    const float* l1_b     = (const float*)d_in[13];
    const float* ln1_w    = (const float*)d_in[14];
    const float* ln1_b    = (const float*)d_in[15];
    const float* adv_uw   = (const float*)d_in[16];
    const float* adv_sw   = (const float*)d_in[17];
    const float* adv_ub   = (const float*)d_in[18];
    const float* adv_sb   = (const float*)d_in[19];
    const float* v_uw     = (const float*)d_in[20];
    const float* v_sw     = (const float*)d_in[21];
    const float* v_ub     = (const float*)d_in[22];
    const float* v_sb     = (const float*)d_in[23];
    const float* adv_rin  = (const float*)d_in[24];
    const float* adv_rout = (const float*)d_in[25];
    const float* v_rin    = (const float*)d_in[26];
    const float* v_rout   = (const float*)d_in[27];

    _Float16* feat16 = (_Float16*)d_ws;
    _Float16* bpack  = (_Float16*)((char*)d_ws + (size_t)B_ * KPAD * sizeof(_Float16));

    graph_feat_kernel<<<B_, 256, 0, stream>>>(
        x, ledge, rows, einfo,
        gcn1_w, gcn1_b, gcn3_w, gcn3_b,
        lng1_w, lng1_b, lng3_w, lng3_b, feat16);

    pack_w_kernel<<<KSTEPS * NTILES, 32, 0, stream>>>(l1_w, bpack);

    gemm_head_kernel<<<B_ / 16, 256, 0, stream>>>(
        feat16, bpack, l1_b, ln1_w, ln1_b,
        adv_uw, adv_sw, adv_ub, adv_sb,
        v_uw, v_sw, v_ub, v_sb,
        adv_rin, adv_rout, v_rin, v_rout,
        (float*)d_out);
}
